// AttentionBlock_72584947303067
// MI455X (gfx1250) — matrix operand order repaired and checked by reading, not compiled, not run on hardware
//
#include <hip/hip_runtime.h>

// -------------------- types for WMMA --------------------
typedef __attribute__((ext_vector_type(16))) __bf16 v16bf;
typedef __attribute__((ext_vector_type(8)))  float  v8f;

union Frag {
    v16bf v;
    unsigned u[8];
};

__device__ __forceinline__ unsigned short f2bf(float f) {
    unsigned u = __builtin_bit_cast(unsigned, f);
    u += 0x7fffu + ((u >> 16) & 1u);     // round-to-nearest-even
    return (unsigned short)(u >> 16);
}

// CDNA5 async global->LDS copy (ASYNCcnt-tracked, no VGPR round trip).
__device__ __forceinline__ void async_b128(unsigned lds_addr, const void* gaddr) {
    asm volatile("global_load_async_to_lds_b128 %0, %1, off"
                 :: "v"(lds_addr), "v"(gaddr) : "memory");
}

// -------------------- GroupNorm --------------------
// x: [B,512,1024] f32 ; out hnT: [B,1024,512] bf16 via LDS transpose tiles.
__global__ __launch_bounds__(256)
void groupnorm_kernel(const float* __restrict__ x,
                      const float* __restrict__ gamma,
                      const float* __restrict__ beta,
                      unsigned short* __restrict__ hnT) {
    const int bg = blockIdx.x;          // b*8 + g
    const int b = bg >> 3, g = bg & 7;
    const size_t base = ((size_t)b * 512 + (size_t)g * 64) * 1024;
    const int tid = threadIdx.x;

    float sum = 0.f, sq = 0.f;
    for (int i = tid; i < 65536; i += 256) {
        float v = x[base + i];
        sum += v; sq += v * v;
    }
    __shared__ float r1[256], r2[256];
    r1[tid] = sum; r2[tid] = sq;
    __syncthreads();
    for (int off = 128; off; off >>= 1) {
        if (tid < off) { r1[tid] += r1[tid + off]; r2[tid] += r2[tid + off]; }
        __syncthreads();
    }
    const float mean = r1[0] * (1.f / 65536.f);
    const float var  = r2[0] * (1.f / 65536.f) - mean * mean;
    const float rstd = rsqrtf(var + 1e-5f);

    __shared__ float sc[64], sh[64];
    if (tid < 64) {
        float ga = gamma[g * 64 + tid];
        sc[tid] = rstd * ga;
        sh[tid] = beta[g * 64 + tid] - mean * rstd * ga;
    }

    __shared__ unsigned short tile[64][65];     // [c][p], padded
    for (int p0 = 0; p0 < 1024; p0 += 64) {
        __syncthreads();
        for (int e = tid; e < 4096; e += 256) {
            int c = e >> 6, p = e & 63;
            float v = x[base + c * 1024 + p0 + p];
            tile[c][p] = f2bf(v * sc[c] + sh[c]);
        }
        __syncthreads();
        // write transposed rows: 64 channels contiguous per pixel
        for (int s = tid; s < 512; s += 256) {
            int p = s >> 3, qd = s & 7;
            unsigned short t[8];
#pragma unroll
            for (int j = 0; j < 8; ++j) t[j] = tile[qd * 8 + j][p];
            uint4 w;
            w.x = (unsigned)t[0] | ((unsigned)t[1] << 16);
            w.y = (unsigned)t[2] | ((unsigned)t[3] << 16);
            w.z = (unsigned)t[4] | ((unsigned)t[5] << 16);
            w.w = (unsigned)t[6] | ((unsigned)t[7] << 16);
            *reinterpret_cast<uint4*>(
                &hnT[((size_t)b * 1024 + p0 + p) * 512 + g * 64 + qd * 8]) = w;
        }
    }
}

// -------------------- weight pack f32 -> n-major K-pair layout --------------------
// wP[o*256 + kp] = { bf16 w[o][2kp] , bf16 w[o][2kp+1] }
__global__ __launch_bounds__(256)
void wpack_kernel(const float* __restrict__ w, unsigned* __restrict__ wP) {
    int j  = blockIdx.x * 256 + threadIdx.x;      // 0 .. 512*256-1
    int o = j >> 8, kp = j & 255;
    float2 f = *reinterpret_cast<const float2*>(&w[o * 512 + 2 * kp]);
    wP[j] = (unsigned)f2bf(f.x) | ((unsigned)f2bf(f.y) << 16);
}

// -------------------- generic batched bf16 WMMA GEMM --------------------
// D[m,n] = sum_k A[m,k] * B[k,n]
// A: bf16 [M,K] row-major.
// BPACK=true : B packed n-major K-pairs, u32[n*ldb + k/2]   (pure async-copy staging)
// BPACK=false: B bf16 [K,N] row-major                       (load + interleave staging)
// MODE 0: out bf16 row-major [M,N], += bias[n] (optional)
// MODE 1: out bf16 transposed [N,M], += bias[n]
// MODE 2: out f32 row-major [M,N], *= scale
// MODE 3: out f32 transposed [N,M], += bias[n] + resid[b][n][m]
// MODE 4: out n-major packed K-pairs u32[n*(M/2) + m/2], += bias[n]
#define BM 128
#define BN 128
#define BK 32
#define LDA_S 40   // padded A row (ushorts): 80B, conflict-free, 16B aligned
#define LDB_S 20   // padded B row (u32s):    80B, conflict-free, 16B aligned

template <int MODE, bool BPACK>
__global__ __launch_bounds__(256)
void gemm_bf16_kernel(const unsigned short* __restrict__ A, size_t sAb, int lda,
                      const void* __restrict__ B, size_t sBb, int ldb,
                      const float* __restrict__ bias,
                      const float* __restrict__ resid, size_t sRb,
                      void* __restrict__ Out, size_t sOb,
                      int M, int N, int K, float scale) {
    __shared__ __align__(16) unsigned short sA[2][BM * LDA_S];  // [row][k] padded
    __shared__ __align__(16) unsigned       sB[2][BN * LDB_S];  // [n][kp] padded

    const int b    = blockIdx.z;
    const int tid  = threadIdx.x;
    const int lane = tid & 31;
    const int wave = tid >> 5;
    const int wm   = wave >> 2;          // 0..1  (64 rows each)
    const int wn   = wave & 3;           // 0..3  (32 cols each)
    const int m0   = blockIdx.x * BM;
    const int n0   = blockIdx.y * BN;
    const int nl   = lane & 15;
    const int hi   = lane >> 4;

    const unsigned short* Ab = A + (size_t)b * sAb;
    const unsigned*       Bp = (const unsigned*)B + (size_t)b * sBb;       // packed
    const unsigned short* Bs = (const unsigned short*)B + (size_t)b * sBb; // raw

    v8f acc[4][2];
#pragma unroll
    for (int mi = 0; mi < 4; ++mi)
#pragma unroll
        for (int ni = 0; ni < 2; ++ni)
            acc[mi][ni] = (v8f)(0.f);

    // branchless single-slot issue helpers (512 slots, 256 threads -> 2 each)
    auto issueA = [&](int bufi, int k0, int s) {
        int row = s >> 2, j = s & 3;
        async_b128((unsigned)(unsigned long long)&sA[bufi][row * LDA_S + j * 8],
                   Ab + (size_t)(m0 + row) * lda + k0 + j * 8);
    };
    auto issueB = [&](int bufi, int k0, int s) {
        int n = s >> 2, j = s & 3;
        async_b128((unsigned)(unsigned long long)&sB[bufi][n * LDB_S + j * 4],
                   Bp + (size_t)(n0 + n) * ldb + k0 / 2 + j * 4);
    };
    auto packB = [&](int bufi, int k0, int s) {
        int kp = s >> 6;
        int n2 = (s & 63) * 2;
        unsigned lo2 = *reinterpret_cast<const unsigned*>(
            Bs + (size_t)(k0 + 2 * kp) * ldb + n0 + n2);
        unsigned hi2 = *reinterpret_cast<const unsigned*>(
            Bs + (size_t)(k0 + 2 * kp + 1) * ldb + n0 + n2);
        sB[bufi][(n2)     * LDB_S + kp] = (lo2 & 0xffffu) | (hi2 << 16);
        sB[bufi][(n2 + 1) * LDB_S + kp] = (lo2 >> 16) | (hi2 & 0xffff0000u);
    };

    auto stage = [&](int bufi, int k0) {
        issueA(bufi, k0, tid);
        issueA(bufi, k0, tid + 256);
        if (BPACK) {
            issueB(bufi, k0, tid);
            issueB(bufi, k0, tid + 256);
        } else {
            packB(bufi, k0, tid);
            packB(bufi, k0, tid + 256);
            packB(bufi, k0, tid + 512);
            packB(bufi, k0, tid + 768);
        }
    };

    auto compute = [&](int cur) {
        Frag af[4], bfr[2];
#pragma unroll
        for (int ni = 0; ni < 2; ++ni) {
            const unsigned* bp = &sB[cur][(wn * 32 + ni * 16 + nl) * LDB_S + hi * 8];
            *reinterpret_cast<uint4*>(&bfr[ni].u[0]) = *reinterpret_cast<const uint4*>(bp);
            *reinterpret_cast<uint4*>(&bfr[ni].u[4]) = *reinterpret_cast<const uint4*>(bp + 4);
        }
#pragma unroll
        for (int mi = 0; mi < 4; ++mi) {
            const unsigned short* rp = &sA[cur][(wm * 64 + mi * 16 + nl) * LDA_S];
#pragma unroll
            for (int v = 0; v < 8; ++v) {
                int k = hi * 16 + v * 2;
                af[mi].u[v] = *reinterpret_cast<const unsigned*>(rp + k);
            }
        }
#pragma unroll
        for (int mi = 0; mi < 4; ++mi)
#pragma unroll
            for (int ni = 0; ni < 2; ++ni)
                acc[mi][ni] = __builtin_amdgcn_wmma_f32_16x16x32_bf16(
                    false, af[mi].v, false, bfr[ni].v,
                    (short)0, acc[mi][ni], false, false);
    };

    const int KT = K / BK;
    stage(0, 0);
    for (int kt = 0; kt < KT - 1; ++kt) {            // steady state: branch-free
        const int cur = kt & 1;
        stage(cur ^ 1, (kt + 1) * BK);
        asm volatile("s_wait_asynccnt %0" :: "n"(BPACK ? 4 : 2) : "memory");
        __syncthreads();
        compute(cur);
        __syncthreads();
    }
    asm volatile("s_wait_asynccnt 0x0" ::: "memory");
    __syncthreads();
    compute((KT - 1) & 1);

    // ---- epilogue: D layout m = tile_m + r + 8*hi, n = tile_n + (lane&15) ----
#pragma unroll
    for (int mi = 0; mi < 4; ++mi) {
#pragma unroll
        for (int ni = 0; ni < 2; ++ni) {
            const int n  = n0 + wn * 32 + ni * 16 + nl;
            const int mb = m0 + wm * 64 + mi * 16 + hi * 8;
            v8f a = acc[mi][ni];
            if (MODE == 0) {
                float bv = bias ? bias[n] : 0.f;
                unsigned short* o = (unsigned short*)Out + (size_t)b * sOb;
#pragma unroll
                for (int r = 0; r < 8; ++r)
                    o[(size_t)(mb + r) * N + n] = f2bf(a[r] + bv);
            } else if (MODE == 1) {
                float bv = bias ? bias[n] : 0.f;
                unsigned short* o = (unsigned short*)Out + (size_t)b * sOb
                                    + (size_t)n * M + mb;
                uint4 pk;
                pk.x = (unsigned)f2bf(a[0] + bv) | ((unsigned)f2bf(a[1] + bv) << 16);
                pk.y = (unsigned)f2bf(a[2] + bv) | ((unsigned)f2bf(a[3] + bv) << 16);
                pk.z = (unsigned)f2bf(a[4] + bv) | ((unsigned)f2bf(a[5] + bv) << 16);
                pk.w = (unsigned)f2bf(a[6] + bv) | ((unsigned)f2bf(a[7] + bv) << 16);
                *reinterpret_cast<uint4*>(o) = pk;
            } else if (MODE == 2) {
                float* o = (float*)Out + (size_t)b * sOb;
#pragma unroll
                for (int r = 0; r < 8; ++r)
                    o[(size_t)(mb + r) * N + n] = a[r] * scale;
            } else if (MODE == 3) {
                const float bv = bias[n];
                const float* rr = resid + (size_t)b * sRb + (size_t)n * M + mb;
                float* o = (float*)Out + (size_t)b * sOb + (size_t)n * M + mb;
                float4 r0 = *reinterpret_cast<const float4*>(rr);
                float4 r1 = *reinterpret_cast<const float4*>(rr + 4);
                float4 o0 = {a[0] + bv + r0.x, a[1] + bv + r0.y,
                             a[2] + bv + r0.z, a[3] + bv + r0.w};
                float4 o1 = {a[4] + bv + r1.x, a[5] + bv + r1.y,
                             a[6] + bv + r1.z, a[7] + bv + r1.w};
                *reinterpret_cast<float4*>(o)     = o0;
                *reinterpret_cast<float4*>(o + 4) = o1;
            } else { // MODE 4: n-major packed K-pairs along m (contiguous uint4)
                float bv = bias ? bias[n] : 0.f;
                unsigned* o32 = (unsigned*)Out + (size_t)b * sOb
                                + (size_t)n * (M / 2) + mb / 2;
                uint4 pk;
                pk.x = (unsigned)f2bf(a[0] + bv) | ((unsigned)f2bf(a[1] + bv) << 16);
                pk.y = (unsigned)f2bf(a[2] + bv) | ((unsigned)f2bf(a[3] + bv) << 16);
                pk.z = (unsigned)f2bf(a[4] + bv) | ((unsigned)f2bf(a[5] + bv) << 16);
                pk.w = (unsigned)f2bf(a[6] + bv) | ((unsigned)f2bf(a[7] + bv) << 16);
                *reinterpret_cast<uint4*>(o32) = pk;
            }
        }
    }
}

// -------------------- softmax (in-place f32 row -> bf16 row) --------------------
__global__ __launch_bounds__(256)
void softmax_rows_kernel(float* __restrict__ S) {
    const int tid = threadIdx.x;
    float* rp = S + (size_t)blockIdx.x * 1024;
    float v0 = rp[tid], v1 = rp[tid + 256], v2 = rp[tid + 512], v3 = rp[tid + 768];

    __shared__ float red[256];
    red[tid] = fmaxf(fmaxf(v0, v1), fmaxf(v2, v3));
    __syncthreads();
    for (int off = 128; off; off >>= 1) {
        if (tid < off) red[tid] = fmaxf(red[tid], red[tid + off]);
        __syncthreads();
    }
    const float mx = red[0];
    __syncthreads();

    v0 = __expf(v0 - mx); v1 = __expf(v1 - mx);
    v2 = __expf(v2 - mx); v3 = __expf(v3 - mx);
    red[tid] = v0 + v1 + v2 + v3;
    __syncthreads();
    for (int off = 128; off; off >>= 1) {
        if (tid < off) red[tid] += red[tid + off];
        __syncthreads();
    }
    const float inv = 1.f / red[0];

    unsigned short* o = (unsigned short*)rp;    // rows already fully read
    o[tid]       = f2bf(v0 * inv);
    o[tid + 256] = f2bf(v1 * inv);
    o[tid + 512] = f2bf(v2 * inv);
    o[tid + 768] = f2bf(v3 * inv);
}

// -------------------- driver --------------------
extern "C" void kernel_launch(void* const* d_in, const int* in_sizes, int n_in,
                              void* d_out, int out_size, void* d_ws, size_t ws_size,
                              hipStream_t stream) {
    const float* x    = (const float*)d_in[0];
    const float* gn_w = (const float*)d_in[1];
    const float* gn_b = (const float*)d_in[2];
    const float* wq   = (const float*)d_in[3];
    const float* bq   = (const float*)d_in[4];
    const float* wk   = (const float*)d_in[5];
    const float* bk   = (const float*)d_in[6];
    const float* wv   = (const float*)d_in[7];
    const float* bv   = (const float*)d_in[8];
    const float* wo   = (const float*)d_in[9];
    const float* bo   = (const float*)d_in[10];
    float* out = (float*)d_out;

    char* ws = (char*)d_ws;
    // workspace layout (bytes)
    unsigned*       wP = (unsigned*)ws;                             //  2 MB: 4x packed 512x256 u32
    unsigned short* hn = (unsigned short*)(ws + 2097152);           // 16 MB: [B,1024,512]
    unsigned short* q  = (unsigned short*)(ws + 18874368);          // 16 MB: [B,1024,512]
    unsigned short* kT = (unsigned short*)(ws + 35651584);          // 16 MB: [B,512,1024]
    unsigned*       vP = (unsigned*)(ws + 52428800);                // 16 MB: [B,512c,512 kp] u32
    float*          s  = (float*)(ws + 69206016);                   // 64 MB: [B,1024,1024]
    unsigned short* ao = hn;                                        // reuse (hn dead after V)

    const size_t WP32 = 512 * 256;                 // one packed weight, u32 elements
    const float scale = 0.04419417382415922f;      // 512^-0.5

    groupnorm_kernel<<<128, 256, 0, stream>>>(x, gn_w, gn_b, hn);
    wpack_kernel<<<512, 256, 0, stream>>>(wq, wP + 0 * WP32);
    wpack_kernel<<<512, 256, 0, stream>>>(wk, wP + 1 * WP32);
    wpack_kernel<<<512, 256, 0, stream>>>(wv, wP + 2 * WP32);
    wpack_kernel<<<512, 256, 0, stream>>>(wo, wP + 3 * WP32);

    dim3 blk(256);
    dim3 g512(8, 4, 16);    // M/128, 512/128, B
    dim3 g1024(8, 8, 16);   // M/128, 1024/128, B

    // Q = hn * WqT + bq            -> bf16 [B,1024,512]
    gemm_bf16_kernel<0, true><<<g512, blk, 0, stream>>>(
        hn, (size_t)1024 * 512, 512, wP + 0 * WP32, 0, 256,
        bq, nullptr, 0, q, (size_t)1024 * 512, 1024, 512, 512, 1.f);
    // K^T = (hn * WkT + bk)^T      -> bf16 [B,512,1024]
    gemm_bf16_kernel<1, true><<<g512, blk, 0, stream>>>(
        hn, (size_t)1024 * 512, 512, wP + 1 * WP32, 0, 256,
        bk, nullptr, 0, kT, (size_t)512 * 1024, 1024, 512, 512, 1.f);
    // V = hn * WvT + bv            -> packed u32 [B, 512 c, 512 kp]
    gemm_bf16_kernel<4, true><<<g512, blk, 0, stream>>>(
        hn, (size_t)1024 * 512, 512, wP + 2 * WP32, 0, 256,
        bv, nullptr, 0, vP, (size_t)512 * 512, 1024, 512, 512, 1.f);
    // S = (Q * K^T) * scale        -> f32 [B,1024,1024]
    gemm_bf16_kernel<2, false><<<g1024, blk, 0, stream>>>(
        q, (size_t)1024 * 512, 512, kT, (size_t)512 * 1024, 1024,
        nullptr, nullptr, 0, s, (size_t)1024 * 1024, 1024, 1024, 512, scale);
    // softmax rows, bf16 in place (rows become lda=2048 bf16)
    softmax_rows_kernel<<<16 * 1024, 256, 0, stream>>>(s);
    // AO = attn * V                -> bf16 [B,1024,512]  (reuses hn buffer)
    gemm_bf16_kernel<0, true><<<g512, blk, 0, stream>>>(
        (const unsigned short*)s, (size_t)1024 * 2048, 2048,
        vP, (size_t)512 * 512, 512,
        nullptr, nullptr, 0, ao, (size_t)1024 * 512, 1024, 512, 1024, 1.f);
    // out = (AO * WoT + bo)^T + x  -> f32 [B,512,1024] = [B,C,H,W]
    gemm_bf16_kernel<3, true><<<g512, blk, 0, stream>>>(
        ao, (size_t)1024 * 512, 512, wP + 3 * WP32, 0, 256,
        bo, x, (size_t)512 * 1024, out, (size_t)512 * 1024, 1024, 512, 512, 1.f);
}
